// CubNet_60481729462415
// MI455X (gfx1250) — compile-verified, matrix operand order checked by reading, not run on hardware
//
#include <hip/hip_runtime.h>
#include <hip/hip_bf16.h>

// ---------------- problem constants ----------------
#define BB   4
#define NN   4096
#define KNN  20
#define EMB  1024
#define ZD   128
#define NC   16
#define AD   64
#define EPS_BN 1e-5f

typedef __attribute__((ext_vector_type(16))) _Float16 v16h;
typedef __attribute__((ext_vector_type(8)))  float    v8f;
typedef _Float16 half_t;

// ---------------- helpers ----------------
__device__ __forceinline__ float lrelu(float x) { return x > 0.f ? x : 0.2f * x; }

__device__ __forceinline__ v8f wmma_f16(v16h a, v16h b, v8f c) {
  // v_wmma_f32_16x16x32_f16  (D = A(16x32) * B(32x16) + C)
  return __builtin_amdgcn_wmma_f32_16x16x32_f16(false, a, false, b, (short)0, c,
                                                false, false);
}

// 16-bit A/B fragment for one 32-wide K-chunk.
// Layout (ISA 7.12.2, 16-bit A 16x32): lane l owns row/col (l&15);
// halves [0..7]  = K (base..base+7), halves [8..15] = K (base+16..base+23),
// base = (l&16) ? 8 : 0.  Two contiguous 8-half runs -> b128 loads.
__device__ __forceinline__ v16h load_frag16(const half_t* __restrict__ p, int lane) {
  const int base = (lane & 16) ? 8 : 0;
  v16h r;
#pragma unroll
  for (int i = 0; i < 8; ++i) r[i] = p[base + i];
#pragma unroll
  for (int i = 0; i < 8; ++i) r[8 + i] = p[base + 16 + i];
  return r;
}

// Same layout but element = xj[k]-xi[k] (edge feature, first half of concat)
__device__ __forceinline__ v16h load_frag16_diff(const half_t* __restrict__ xj,
                                                 const half_t* __restrict__ xi,
                                                 int kOff, int lane) {
  const int base = (lane & 16) ? 8 : 0;
  v16h r;
#pragma unroll
  for (int i = 0; i < 8; ++i) {
    int k = kOff + base + i;
    r[i] = (half_t)((float)xj[k] - (float)xi[k]);
  }
#pragma unroll
  for (int i = 0; i < 8; ++i) {
    int k = kOff + base + 16 + i;
    r[8 + i] = (half_t)((float)xj[k] - (float)xi[k]);
  }
  return r;
}

// store C-fragment (32-bit C/D layout: lane owns col (l&15), rows v + 8*(l>>4))
__device__ __forceinline__ void store_tile(half_t* __restrict__ out, int rowBase,
                                           int colBase, int ldo, v8f c, int lane) {
  const int col = colBase + (lane & 15);
  const int rh  = (lane >> 4) & 1;
#pragma unroll
  for (int v = 0; v < 8; ++v)
    out[(size_t)(rowBase + rh * 8 + v) * ldo + col] = (half_t)c[v];
}

// per-out-channel sum / sumsq accumulation (BN training stats)
__device__ __forceinline__ void stats_accum(v8f c, int colBase, int lane,
                                            float* __restrict__ stats) {
  float s = 0.f, s2 = 0.f;
#pragma unroll
  for (int v = 0; v < 8; ++v) { s += c[v]; s2 += c[v] * c[v]; }
  s  += __shfl_xor(s, 16);
  s2 += __shfl_xor(s2, 16);
  if (lane < 16) {
    atomicAdd(&stats[2 * (colBase + lane) + 0], s);
    atomicAdd(&stats[2 * (colBase + lane) + 1], s2);
  }
}

// LDS-resident top-K insert: register 'bound' guards the rare slow path so the
// hot loop carries ONE live VGPR instead of 40 (avoids spills around v_wmma).
__device__ __forceinline__ void topk_lds_insert(float* __restrict__ dl,
                                                int* __restrict__ il,
                                                float& bound, float d, int i) {
  if (__builtin_expect(d < bound, 0)) {
    int j = KNN - 1;
    while (j > 0 && dl[j - 1] > d) {
      dl[j] = dl[j - 1];
      il[j] = il[j - 1];
      --j;
    }
    dl[j] = d;
    il[j] = i;
    bound = dl[KNN - 1];
  }
}

// ---------------- utility kernels ----------------
__global__ void f32_to_f16_kernel(const float* __restrict__ in,
                                  half_t* __restrict__ out, int n) {
  int i = blockIdx.x * 256 + threadIdx.x;
  if (i < n) out[i] = (half_t)in[i];
}

__global__ void sqnorm_kernel(const float* __restrict__ x, float* __restrict__ xx,
                              int C) {
  int i = blockIdx.x * 256 + threadIdx.x;   // i < BB*NN
  const float* p = x + (size_t)i * C;
  float s = 0.f;
  for (int c = 0; c < C; ++c) s += p[c] * p[c];
  xx[i] = s;
}

// ---------------- knn on xyz (C=3, VALU; top-k lists in LDS) ----------------
__global__ void knn3_kernel(const float* __restrict__ xyz, int* __restrict__ idx) {
  __shared__ float sx[256 * 3];
  __shared__ float sdist[256][KNN];
  __shared__ int   sidx[256][KNN];
  const int b = blockIdx.y;
  const int n = blockIdx.x * 256 + threadIdx.x;
  const float* base = xyz + (size_t)b * NN * 3;
  const float xi0 = base[n * 3 + 0], xi1 = base[n * 3 + 1], xi2 = base[n * 3 + 2];
  float* dl = sdist[threadIdx.x];
  int*   il = sidx[threadIdx.x];
  for (int k = 0; k < KNN; ++k) { dl[k] = 3.4e38f; il[k] = 0; }
  float bound = 3.4e38f;
  for (int c0 = 0; c0 < NN; c0 += 256) {
    sx[threadIdx.x * 3 + 0] = base[(c0 + threadIdx.x) * 3 + 0];
    sx[threadIdx.x * 3 + 1] = base[(c0 + threadIdx.x) * 3 + 1];
    sx[threadIdx.x * 3 + 2] = base[(c0 + threadIdx.x) * 3 + 2];
    __syncthreads();
    for (int t = 0; t < 256; ++t) {
      float dx = sx[t * 3 + 0] - xi0;
      float dy = sx[t * 3 + 1] - xi1;
      float dz = sx[t * 3 + 2] - xi2;
      topk_lds_insert(dl, il, bound, dx * dx + dy * dy + dz * dz, c0 + t);
    }
    __syncthreads();
  }
  size_t o = ((size_t)b * NN + n) * KNN;
  for (int k = 0; k < KNN; ++k) idx[o + k] = il[k];
}

// ---------------- knn on x1 (C=64, WMMA Gram matrix) ----------------
__global__ void knn64_wmma_kernel(const half_t* __restrict__ xh,
                                  const float* __restrict__ xx,
                                  int* __restrict__ idx) {
  __shared__ float tile[8][16][16];
  __shared__ float sdist[8][16][KNN];
  __shared__ int   sidx[8][16][KNN];
  const int wave = threadIdx.x >> 5, lane = threadIdx.x & 31;
  const int b = blockIdx.y;
  const int r0 = blockIdx.x * 128 + wave * 16;
  const int row = r0 + (lane & 15);
  if (lane < 16) {
    for (int k = 0; k < KNN; ++k) {
      sdist[wave][lane][k] = 3.4e38f;
      sidx[wave][lane][k] = 0;
    }
  }
  const half_t* arow = xh + ((size_t)b * NN + row) * 64;
  const v16h a0 = load_frag16(arow, lane);
  const v16h a1 = load_frag16(arow + 32, lane);
  float bound = 3.4e38f;
  for (int c0 = 0; c0 < NN; c0 += 16) {
    const int cand = c0 + (lane & 15);
    const half_t* brow = xh + ((size_t)b * NN + cand) * 64;
    if (c0 + 16 < NN)
      __builtin_prefetch(xh + ((size_t)b * NN + cand + 16) * 64, 0, 1);
    v16h b0 = load_frag16(brow, lane);
    v16h b1 = load_frag16(brow + 32, lane);
    v8f c = {};
    c = wmma_f16(a0, b0, c);
    c = wmma_f16(a1, b1, c);
    // d2 - const = xx[cand] - 2*inner   (row term is constant -> order-invariant)
    const float xxc = xx[b * NN + cand];
    const int col = lane & 15, rh = (lane >> 4) & 1;
#pragma unroll
    for (int v = 0; v < 8; ++v) tile[wave][rh * 8 + v][col] = xxc - 2.0f * c[v];
    __syncthreads();
    if (lane < 16) {
      float* dl = sdist[wave][lane];
      int*   il = sidx[wave][lane];
#pragma unroll
      for (int cc = 0; cc < 16; ++cc)
        topk_lds_insert(dl, il, bound, tile[wave][lane][cc], c0 + cc);
    }
    __syncthreads();
  }
  if (lane < 16) {
    size_t o = ((size_t)b * NN + row) * KNN;
    for (int k = 0; k < KNN; ++k) idx[o + k] = sidx[wave][lane][k];
  }
}

// ---------------- stage-1 conv a: edge feature (6) -> 64, VALU + stats --------
__global__ void conv_feat6_stats(const float* __restrict__ xyz,
                                 const int* __restrict__ idx0,
                                 const float* __restrict__ w1a,
                                 half_t* __restrict__ h, float* __restrict__ stats) {
  __shared__ float w[64 * 6];
  __shared__ float ls[128];
  for (int i = threadIdx.x; i < 64 * 6; i += 256) w[i] = w1a[i];
  if (threadIdx.x < 128) ls[threadIdx.x] = 0.f;
  __syncthreads();
  const int r = blockIdx.x * 256 + threadIdx.x;       // r < BB*NN*KNN
  const int b = r / (NN * KNN);
  const int rem = r % (NN * KNN);
  const int n = rem / KNN;
  const int j = idx0[r];
  const float* pi = xyz + ((size_t)b * NN + n) * 3;
  const float* pj = xyz + ((size_t)b * NN + j) * 3;
  float f[6] = { pj[0] - pi[0], pj[1] - pi[1], pj[2] - pi[2], pi[0], pi[1], pi[2] };
  half_t* ho = h + (size_t)r * 64;
  for (int o = 0; o < 64; ++o) {
    float acc = 0.f;
#pragma unroll
    for (int c = 0; c < 6; ++c) acc += f[c] * w[o * 6 + c];
    ho[o] = (half_t)acc;
    atomicAdd(&ls[o], acc);
    atomicAdd(&ls[64 + o], acc * acc);
  }
  __syncthreads();
  if (threadIdx.x < 64) {
    atomicAdd(&stats[2 * threadIdx.x + 0], ls[threadIdx.x]);
    atomicAdd(&stats[2 * threadIdx.x + 1], ls[64 + threadIdx.x]);
  }
}

// ---------------- BN(lrelu) elementwise over [rows,64] f16 in-place ----------
__global__ void bn_lrelu_inplace(half_t* __restrict__ h,
                                 const float* __restrict__ stats, float inv_cnt) {
  size_t i = (size_t)blockIdx.x * 256 + threadIdx.x;
  int c = (int)(i & 63);
  float m  = stats[2 * c] * inv_cnt;
  float vv = stats[2 * c + 1] * inv_cnt - m * m;
  float rs = rsqrtf(vv + EPS_BN);
  float x = ((float)h[i] - m) * rs;
  h[i] = (half_t)lrelu(x);
}

// ---------------- generic 64->64 WMMA GEMM, in-place, with stats -------------
__global__ void gemm64_inplace_stats(half_t* __restrict__ h,
                                     const half_t* __restrict__ w,
                                     float* __restrict__ stats) {
  const int wave = threadIdx.x >> 5, lane = threadIdx.x & 31;
  const int rowTile = blockIdx.x * 8 + wave;
  const int r = rowTile * 16 + (lane & 15);
  const half_t* arow = h + (size_t)r * 64;
  const v16h a0 = load_frag16(arow, lane);
  const v16h a1 = load_frag16(arow + 32, lane);
  v8f acc[4];
#pragma unroll
  for (int ct = 0; ct < 4; ++ct) {
    const int col = ct * 16 + (lane & 15);
    const half_t* wrow = w + (size_t)col * 64;
    v16h b0 = load_frag16(wrow, lane);
    v16h b1 = load_frag16(wrow + 32, lane);
    v8f c = {};
    c = wmma_f16(a0, b0, c);
    c = wmma_f16(a1, b1, c);
    acc[ct] = c;
  }
#pragma unroll
  for (int ct = 0; ct < 4; ++ct) {
    store_tile(h, rowTile * 16, ct * 16, 64, acc[ct], lane);
    stats_accum(acc[ct], ct * 16, lane, stats);
  }
}

// ---------------- BN + lrelu + max over k -> x (f32 and f16) -----------------
__global__ void bn_lrelu_maxk(const half_t* __restrict__ h,
                              const float* __restrict__ stats, float inv_cnt,
                              float* __restrict__ xf, half_t* __restrict__ xhh) {
  const int t = blockIdx.x * 256 + threadIdx.x;   // BB*NN*64
  const int c = t & 63, bn = t >> 6;
  float m  = stats[2 * c] * inv_cnt;
  float vv = stats[2 * c + 1] * inv_cnt - m * m;
  float rs = rsqrtf(vv + EPS_BN);
  const half_t* p = h + (size_t)bn * KNN * 64 + c;
  float mx = -3.4e38f;
  for (int kk = 0; kk < KNN; ++kk) {
    float x = ((float)p[kk * 64] - m) * rs;
    mx = fmaxf(mx, lrelu(x));
  }
  xf[t] = mx;
  xhh[t] = (half_t)mx;
}

// ---------------- stage-2 conv a: edge feature (128) -> 64, WMMA -------------
__global__ void gemm_feat128_stats(const half_t* __restrict__ x1h,
                                   const int* __restrict__ idx1,
                                   const half_t* __restrict__ w,
                                   half_t* __restrict__ hout,
                                   float* __restrict__ stats) {
  const int wave = threadIdx.x >> 5, lane = threadIdx.x & 31;
  const int rowTile = blockIdx.x * 8 + wave;
  const int r = rowTile * 16 + (lane & 15);
  const int b = r / (NN * KNN);
  const int rem = r % (NN * KNN);
  const int n = rem / KNN;
  const int j = idx1[r];
  const half_t* xi = x1h + ((size_t)b * NN + n) * 64;
  const half_t* xj = x1h + ((size_t)b * NN + j) * 64;
  v16h a[4];
  a[0] = load_frag16_diff(xj, xi, 0, lane);    // channels 0..31  : xj-xi
  a[1] = load_frag16_diff(xj, xi, 32, lane);   // channels 32..63 : xj-xi
  a[2] = load_frag16(xi, lane);                // channels 64..95 : xi
  a[3] = load_frag16(xi + 32, lane);           // channels 96..127: xi
#pragma unroll
  for (int ct = 0; ct < 4; ++ct) {
    const int col = ct * 16 + (lane & 15);
    const half_t* wrow = w + (size_t)col * 128;
    v8f c = {};
#pragma unroll
    for (int kc = 0; kc < 4; ++kc)
      c = wmma_f16(a[kc], load_frag16(wrow + kc * 32, lane), c);
    store_tile(hout, rowTile * 16, ct * 16, 64, c, lane);
    stats_accum(c, ct * 16, lane, stats);
  }
}

// ---------------- concat x1|x2 -> x_per (f16) --------------------------------
__global__ void concat_xper(const half_t* __restrict__ x1h,
                            const half_t* __restrict__ x2h,
                            half_t* __restrict__ xp) {
  const int i = blockIdx.x * 256 + threadIdx.x;   // BB*NN*128
  const int bn = i >> 7, c = i & 127;
  xp[i] = (c < 64) ? x1h[(size_t)bn * 64 + c] : x2h[(size_t)bn * 64 + (c - 64)];
}

// ---------------- global feature pass A: WMMA stats --------------------------
__global__ void gemm_g_stats(const half_t* __restrict__ xp,
                             const half_t* __restrict__ w3h,
                             float* __restrict__ stats) {
  const int wave = threadIdx.x >> 5, lane = threadIdx.x & 31;
  const int task = blockIdx.x * 8 + wave;           // 65536 tasks
  const int rowTile = task >> 6, colTile = task & 63;
  const int r = rowTile * 16 + (lane & 15);
  const half_t* arow = xp + (size_t)r * 128;
  const int col = colTile * 16 + (lane & 15);
  const half_t* wrow = w3h + (size_t)col * 128;
  v8f c = {};
#pragma unroll
  for (int kc = 0; kc < 4; ++kc)
    c = wmma_f16(load_frag16(arow + kc * 32, lane),
                 load_frag16(wrow + kc * 32, lane), c);
  stats_accum(c, colTile * 16, lane, stats);
}

// ---------------- global feature pass B: WMMA recompute + BN + lrelu + max ---
__global__ void gemm_g_max(const half_t* __restrict__ xp,
                           const half_t* __restrict__ w3h,
                           const float* __restrict__ stats,
                           float* __restrict__ xg) {
  const int wave = threadIdx.x >> 5, lane = threadIdx.x & 31;
  const int task = blockIdx.x * 8 + wave;           // BB*64 tasks
  const int b = task >> 6, colTile = task & 63;
  const int col = colTile * 16 + (lane & 15);
  const half_t* wrow = w3h + (size_t)col * 128;
  v16h bf[4];
#pragma unroll
  for (int kc = 0; kc < 4; ++kc) bf[kc] = load_frag16(wrow + kc * 32, lane);
  const float inv = 1.0f / (float)(BB * NN);
  const float m  = stats[2 * col] * inv;
  const float rs = rsqrtf(stats[2 * col + 1] * inv - m * m + EPS_BN);
  float mx = -3.4e38f;
  for (int rt = 0; rt < NN / 16; ++rt) {
    const int r = b * NN + rt * 16 + (lane & 15);
    const half_t* arow = xp + (size_t)r * 128;
    v8f c = {};
#pragma unroll
    for (int kc = 0; kc < 4; ++kc)
      c = wmma_f16(load_frag16(arow + kc * 32, lane), bf[kc], c);
#pragma unroll
    for (int v = 0; v < 8; ++v) mx = fmaxf(mx, lrelu((c[v] - m) * rs));
  }
  mx = fmaxf(mx, __shfl_xor(mx, 16));
  if (lane < 16) xg[b * EMB + col] = mx;
}

// ---------------- head: mu / log_var / z -------------------------------------
__global__ void head_mvz(const float* __restrict__ xg,
                         const float* __restrict__ mu_w, const float* __restrict__ mu_b,
                         const float* __restrict__ var_w, const float* __restrict__ var_b,
                         const float* __restrict__ eps,
                         float* __restrict__ out_mu, float* __restrict__ out_lv,
                         float* __restrict__ z) {
  const int t = blockIdx.x * 128 + threadIdx.x;   // BB*ZD
  const int b = t >> 7, o = t & 127;
  const float* g = xg + b * EMB;
  float sm = 0.f, sv = 0.f;
  for (int c = 0; c < EMB; ++c) {
    sm += g[c] * mu_w[(size_t)o * EMB + c];
    sv += g[c] * var_w[(size_t)o * EMB + c];
  }
  const float mu = sm + mu_b[o], lv = sv + var_b[o];
  out_mu[t] = mu;
  out_lv[t] = lv;
  z[t] = eps[t] * expf(0.5f * lv) + mu;
}

// ---------------- cuboid decoder layer 1: (z|lrelu(enc)) @ cub_w1 ------------
__global__ void cub1_kernel(const float* __restrict__ z,
                            const float* __restrict__ enc_w,
                            const float* __restrict__ w1,
                            float* __restrict__ xc1) {
  const int t = blockIdx.x * 256 + threadIdx.x;   // BB*256*NC
  const int b = t / (256 * NC);
  const int rem = t % (256 * NC);
  const int o = rem / NC, m = rem % NC;
  float acc = 0.f;
  for (int c = 0; c < ZD; ++c) acc += z[b * ZD + c] * w1[o * (ZD + 64) + c];
  for (int c = 0; c < 64; ++c) {
    float e = lrelu(enc_w[c * NC + m]);
    acc += e * w1[o * (ZD + 64) + ZD + c];
  }
  xc1[t] = lrelu(acc);
}

// ---------------- cuboid decoder layer 2 -------------------------------------
__global__ void cub2_kernel(const float* __restrict__ xc1,
                            const float* __restrict__ w2,
                            float* __restrict__ xcub) {
  const int t = blockIdx.x * 256 + threadIdx.x;   // BB*128*NC
  const int b = t / (128 * NC);
  const int rem = t % (128 * NC);
  const int o = rem / NC, m = rem % NC;
  float acc = 0.f;
  for (int c = 0; c < 256; ++c) acc += xc1[(b * 256 + c) * NC + m] * w2[o * 256 + c];
  xcub[t] = lrelu(acc);
}

// ---------------- Para_pred + K projection -----------------------------------
__global__ void para_pred_kernel(const float* __restrict__ xcub,
                                 const float* __restrict__ rot_w, const float* __restrict__ rot_b,
                                 const float* __restrict__ trans_w, const float* __restrict__ trans_b,
                                 const float* __restrict__ ext_w1, const float* __restrict__ ext_b1,
                                 const float* __restrict__ ext_w2, const float* __restrict__ ext_b2,
                                 const float* __restrict__ k_w,
                                 float* __restrict__ rotate, float* __restrict__ trans,
                                 float* __restrict__ exist, float* __restrict__ Km) {
  const int t = threadIdx.x;   // 64 threads, one block
  const int b = t >> 4, m = t & 15;
  float x[128];
  for (int c = 0; c < 128; ++c) x[c] = xcub[(b * 128 + c) * NC + m];
  // quaternion
  float q[4];
#pragma unroll
  for (int o = 0; o < 4; ++o) {
    float a = rot_b[o];
    for (int c = 0; c < 128; ++c) a += x[c] * rot_w[o * 128 + c];
    q[o] = a;
  }
  float nrm = sqrtf(q[0] * q[0] + q[1] * q[1] + q[2] * q[2] + q[3] * q[3]);
  nrm = fmaxf(nrm, 1e-12f);
  const float w = q[0] / nrm, xq = q[1] / nrm, yq = q[2] / nrm, zq = q[3] / nrm;
  const float xx = xq * xq, yy = yq * yq, zz = zq * zq;
  float* R = rotate + (b * NC + m) * 9;
  R[0] = 1.f - 2.f * (yy + zz); R[1] = 2.f * (xq * yq - w * zq); R[2] = 2.f * (xq * zq + w * yq);
  R[3] = 2.f * (xq * yq + w * zq); R[4] = 1.f - 2.f * (xx + zz); R[5] = 2.f * (yq * zq - w * xq);
  R[6] = 2.f * (xq * zq - w * yq); R[7] = 2.f * (yq * zq + w * xq); R[8] = 1.f - 2.f * (xx + yy);
  // translation
#pragma unroll
  for (int o = 0; o < 3; ++o) {
    float a = trans_b[o];
    for (int c = 0; c < 128; ++c) a += x[c] * trans_w[o * 128 + c];
    trans[(b * NC + m) * 3 + o] = tanhf(a);
  }
  // existence
  float ex = ext_b2[0];
  for (int o = 0; o < 30; ++o) {
    float a = ext_b1[o];
    for (int c = 0; c < 128; ++c) a += x[c] * ext_w1[o * 128 + c];
    ex += lrelu(a) * ext_w2[o];
  }
  exist[b * NC + m] = ex;
  // Km = k_w @ x_cuboid[:, m]
  for (int o = 0; o < AD; ++o) {
    float a = 0.f;
    for (int c = 0; c < 128; ++c) a += x[c] * k_w[o * 128 + c];
    Km[(b * NC + m) * AD + o] = a;
  }
}

// ---------------- attention + output assembly --------------------------------
__global__ void attention_kernel(const float* __restrict__ x1f,
                                 const float* __restrict__ x2f,
                                 const float* __restrict__ q_w,
                                 const float* __restrict__ Km,
                                 const float* __restrict__ rotate,
                                 const float* __restrict__ trans,
                                 float* __restrict__ out_assign,
                                 float* __restrict__ out_rot,
                                 float* __restrict__ out_trans) {
  __shared__ float sqw[AD * 128];
  __shared__ float sKm[NC * AD];
  __shared__ float sR[NC * 9];
  __shared__ float sT[NC * 3];
  const int b = blockIdx.y;
  const int n = blockIdx.x * 256 + threadIdx.x;
  for (int i = threadIdx.x; i < AD * 128; i += 256) sqw[i] = q_w[i];
  for (int i = threadIdx.x; i < NC * AD; i += 256) sKm[i] = Km[b * NC * AD + i];
  if (threadIdx.x < NC * 9) sR[threadIdx.x] = rotate[b * NC * 9 + threadIdx.x];
  if (threadIdx.x < NC * 3) sT[threadIdx.x] = trans[b * NC * 3 + threadIdx.x];
  __syncthreads();
  float x[128];
  const size_t bn = (size_t)b * NN + n;
  for (int c = 0; c < 64; ++c) { x[c] = x1f[bn * 64 + c]; x[64 + c] = x2f[bn * 64 + c]; }
  float Q[AD];
  for (int o = 0; o < AD; ++o) {
    float a = 0.f;
    for (int c = 0; c < 128; ++c) a += x[c] * sqw[o * 128 + c];
    Q[o] = a * 0.125f;   // 1/sqrt(AD)
  }
  float sc[NC];
  float mx = -3.4e38f;
#pragma unroll
  for (int m = 0; m < NC; ++m) {
    float a = 0.f;
    for (int d = 0; d < AD; ++d) a += sKm[m * AD + d] * Q[d];
    sc[m] = a;
    mx = fmaxf(mx, a);
  }
  float ssum = 0.f;
#pragma unroll
  for (int m = 0; m < NC; ++m) { sc[m] = expf(sc[m] - mx); ssum += sc[m]; }
  const float inv = 1.f / ssum;
#pragma unroll
  for (int m = 0; m < NC; ++m) {
    sc[m] *= inv;
    out_assign[bn * NC + m] = sc[m];
  }
#pragma unroll
  for (int ij = 0; ij < 9; ++ij) {
    float a = 0.f;
#pragma unroll
    for (int m = 0; m < NC; ++m) a += sc[m] * sR[m * 9 + ij];
    out_rot[bn * 9 + ij] = a;
  }
#pragma unroll
  for (int k = 0; k < 3; ++k) {
    float a = 0.f;
#pragma unroll
    for (int m = 0; m < NC; ++m) a += sc[m] * sT[m * 3 + k];
    out_trans[bn * 3 + k] = a;
  }
}

// ---------------- launch orchestration ---------------------------------------
static inline void* carve(void* base, size_t& cur, size_t bytes) {
  cur = (cur + 255) & ~(size_t)255;
  void* p = (char*)base + cur;
  cur += bytes;
  return p;
}

extern "C" void kernel_launch(void* const* d_in, const int* in_sizes, int n_in,
                              void* d_out, int out_size, void* d_ws, size_t ws_size,
                              hipStream_t stream) {
  (void)in_sizes; (void)n_in; (void)out_size; (void)ws_size;
  const float* xyz    = (const float*)d_in[0];
  const float* eps_z  = (const float*)d_in[2];
  const float* w1a    = (const float*)d_in[3];
  const float* w1b    = (const float*)d_in[4];
  const float* w2a    = (const float*)d_in[5];
  const float* w2b    = (const float*)d_in[6];
  const float* w3     = (const float*)d_in[7];
  const float* mu_w   = (const float*)d_in[8];
  const float* mu_b   = (const float*)d_in[9];
  const float* var_w  = (const float*)d_in[10];
  const float* var_b  = (const float*)d_in[11];
  const float* enc_w  = (const float*)d_in[12];
  const float* cub_w1 = (const float*)d_in[13];
  const float* cub_w2 = (const float*)d_in[14];
  const float* q_w    = (const float*)d_in[15];
  const float* k_w    = (const float*)d_in[16];
  const float* rot_w  = (const float*)d_in[19];
  const float* rot_b  = (const float*)d_in[20];
  const float* trans_w= (const float*)d_in[21];
  const float* trans_b= (const float*)d_in[22];
  const float* ext_w1 = (const float*)d_in[23];
  const float* ext_b1 = (const float*)d_in[24];
  const float* ext_w2 = (const float*)d_in[25];
  const float* ext_b2 = (const float*)d_in[26];

  float* out = (float*)d_out;
  float* out_assign = out;                // [B,N,NC]     262144
  float* out_rot    = out + 262144;       // [B,N,3,3]    147456
  float* out_trans  = out + 409600;       // [B,N,3]       49152
  float* out_exist  = out + 458752;       // [B,NC,1]         64
  float* out_mu     = out + 458816;       // [B,ZD]          512
  float* out_lv     = out + 459328;       // [B,ZD]          512

  size_t cur = 0;
  const size_t ROWS = (size_t)BB * NN * KNN;                 // 327680
  half_t* w1bh  = (half_t*)carve(d_ws, cur, 64 * 64 * 2);
  half_t* w2ah  = (half_t*)carve(d_ws, cur, 64 * 128 * 2);
  half_t* w2bh  = (half_t*)carve(d_ws, cur, 64 * 64 * 2);
  half_t* w3h   = (half_t*)carve(d_ws, cur, (size_t)EMB * 128 * 2);
  float*  xx1   = (float*)carve(d_ws, cur, (size_t)BB * NN * 4);
  int*    idx0  = (int*)carve(d_ws, cur, ROWS * 4);
  int*    idx1  = (int*)carve(d_ws, cur, ROWS * 4);
  half_t* h     = (half_t*)carve(d_ws, cur, ROWS * 64 * 2);  // 40 MB, reused
  float*  stats = (float*)carve(d_ws, cur, 2048 * 2 * 4);
  float*  x1f   = (float*)carve(d_ws, cur, (size_t)BB * NN * 64 * 4);
  half_t* x1h   = (half_t*)carve(d_ws, cur, (size_t)BB * NN * 64 * 2);
  float*  x2f   = (float*)carve(d_ws, cur, (size_t)BB * NN * 64 * 4);
  half_t* x2h   = (half_t*)carve(d_ws, cur, (size_t)BB * NN * 64 * 2);
  half_t* xperh = (half_t*)carve(d_ws, cur, (size_t)BB * NN * 128 * 2);
  float*  xg    = (float*)carve(d_ws, cur, (size_t)BB * EMB * 4);
  float*  zbuf  = (float*)carve(d_ws, cur, (size_t)BB * ZD * 4);
  float*  xc1   = (float*)carve(d_ws, cur, (size_t)BB * 256 * NC * 4);
  float*  xcub  = (float*)carve(d_ws, cur, (size_t)BB * 128 * NC * 4);
  float*  rotbuf= (float*)carve(d_ws, cur, (size_t)BB * NC * 9 * 4);
  float*  trbuf = (float*)carve(d_ws, cur, (size_t)BB * NC * 3 * 4);
  float*  KmBuf = (float*)carve(d_ws, cur, (size_t)BB * NC * AD * 4);

  const float invR = 1.0f / (float)ROWS;

  // weights -> f16
  f32_to_f16_kernel<<<(64 * 64 + 255) / 256, 256, 0, stream>>>(w1b, w1bh, 64 * 64);
  f32_to_f16_kernel<<<(64 * 128 + 255) / 256, 256, 0, stream>>>(w2a, w2ah, 64 * 128);
  f32_to_f16_kernel<<<(64 * 64 + 255) / 256, 256, 0, stream>>>(w2b, w2bh, 64 * 64);
  f32_to_f16_kernel<<<(EMB * 128 + 255) / 256, 256, 0, stream>>>(w3, w3h, EMB * 128);

  // ---- stage 1 ----
  knn3_kernel<<<dim3(NN / 256, BB), 256, 0, stream>>>(xyz, idx0);
  hipMemsetAsync(stats, 0, 4096 * 4, stream);
  conv_feat6_stats<<<ROWS / 256, 256, 0, stream>>>(xyz, idx0, w1a, h, stats);
  bn_lrelu_inplace<<<(unsigned)(ROWS * 64 / 256), 256, 0, stream>>>(h, stats, invR);
  hipMemsetAsync(stats, 0, 4096 * 4, stream);
  gemm64_inplace_stats<<<(unsigned)(ROWS / 16 / 8), 256, 0, stream>>>(h, w1bh, stats);
  bn_lrelu_maxk<<<BB * NN * 64 / 256, 256, 0, stream>>>(h, stats, invR, x1f, x1h);

  // ---- stage 2 ----
  sqnorm_kernel<<<BB * NN / 256, 256, 0, stream>>>(x1f, xx1, 64);
  knn64_wmma_kernel<<<dim3(NN / 128, BB), 256, 0, stream>>>(x1h, xx1, idx1);
  hipMemsetAsync(stats, 0, 4096 * 4, stream);
  gemm_feat128_stats<<<(unsigned)(ROWS / 16 / 8), 256, 0, stream>>>(x1h, idx1, w2ah, h, stats);
  bn_lrelu_inplace<<<(unsigned)(ROWS * 64 / 256), 256, 0, stream>>>(h, stats, invR);
  hipMemsetAsync(stats, 0, 4096 * 4, stream);
  gemm64_inplace_stats<<<(unsigned)(ROWS / 16 / 8), 256, 0, stream>>>(h, w2bh, stats);
  bn_lrelu_maxk<<<BB * NN * 64 / 256, 256, 0, stream>>>(h, stats, invR, x2f, x2h);

  // ---- global feature ----
  concat_xper<<<BB * NN * 128 / 256, 256, 0, stream>>>(x1h, x2h, xperh);
  hipMemsetAsync(stats, 0, 4096 * 4, stream);
  gemm_g_stats<<<(BB * NN / 16) * (EMB / 16) / 8, 256, 0, stream>>>(xperh, w3h, stats);
  gemm_g_max<<<BB * (EMB / 16) / 8, 256, 0, stream>>>(xperh, w3h, stats, xg);

  // ---- head ----
  head_mvz<<<BB, 128, 0, stream>>>(xg, mu_w, mu_b, var_w, var_b, eps_z,
                                   out_mu, out_lv, zbuf);
  cub1_kernel<<<BB * 256 * NC / 256, 256, 0, stream>>>(zbuf, enc_w, cub_w1, xc1);
  cub2_kernel<<<BB * 128 * NC / 256, 256, 0, stream>>>(xc1, cub_w2, xcub);
  para_pred_kernel<<<1, 64, 0, stream>>>(xcub, rot_w, rot_b, trans_w, trans_b,
                                         ext_w1, ext_b1, ext_w2, ext_b2, k_w,
                                         rotbuf, trbuf, out_exist, KmBuf);

  // ---- attention + outputs ----
  attention_kernel<<<dim3(NN / 256, BB), 256, 0, stream>>>(
      x1f, x2f, q_w, KmBuf, rotbuf, trbuf, out_assign, out_rot, out_trans);
}
